// UltraLight_VM_UNet_11639361372353
// MI455X (gfx1250) — matrix operand order repaired and checked by reading, not compiled, not run on hardware
//
// UltraLight VM-UNet forward for MI455X (gfx1250), round 3.
// Changes vs round 2:
//  * WMMA GEMM v3: the packed-weight B panel is now staged into LDS by the
//    Tensor Data Mover -- one tensor_load_to_lds per workgroup (D# built on
//    host, group0/group1 in SGPRs), completion via s_wait_tensorcnt 0.
//    The bias vector is staged via GLOBAL_LOAD_ASYNC_TO_LDS_B32 (ASYNCcnt),
//    so both gfx1250 async paths are exercised where they genuinely help.
//  * A fragments remain direct global_load_b128 + global_prefetch_b8 of the
//    next K-strip; B fragments come from LDS (ds_load_b128) and each A
//    fragment is reused across up to 4 N tiles / WMMA accumulators.
// Rationale: the network is bandwidth-bound (~3 GFLOP vs ~0.5 GB staged
// traffic at 23.3 TB/s); A traffic dominates (up to 2M rows), B panels are
// tiny (<=10 KB) and belong in LDS via TDM, not in per-lane load streams.

#include <hip/hip_runtime.h>
#include <hip/hip_fp16.h>

#define CDIV(a, b) (((a) + (b) - 1) / (b))

typedef __attribute__((ext_vector_type(16))) _Float16 v16h;
typedef __attribute__((ext_vector_type(8)))  float    v8f;
typedef __attribute__((ext_vector_type(4)))  unsigned u32x4;
typedef __attribute__((ext_vector_type(8)))  unsigned u32x8;

__device__ __forceinline__ float gelu_f(float x) { return 0.5f * x * (1.0f + erff(x * 0.70710678118f)); }
__device__ __forceinline__ float silu_f(float x) { return x / (1.0f + __expf(-x)); }
__device__ __forceinline__ float sigm_f(float x) { return 1.0f / (1.0f + __expf(-x)); }
__device__ __forceinline__ float softplus_f(float x) { return (x > 20.0f) ? x : log1pf(__expf(x)); }

// Tensor DMA descriptor (D#) groups 0+1, built host-side; d[1] (lds_addr)
// patched in-kernel with the LDS offset of the staging buffer.
struct TDesc { unsigned d[12]; };

// ---------------------------------------------------------------------------
// WMMA GEMM v3: C[M,Np] = A[M,Kp](f16 row-major) * Bp[Np,Kp]^T(f16 n-major)
// + bias + activation.
//  - B panel (Np*Kp <= 8192 halfs) DMA'd into LDS by the TDM (TENSORcnt)
//  - bias staged into LDS via async-to-LDS (ASYNCcnt)
//  - 8 waves/block, each wave owns one M tile, sweeps up to 4 N tiles
// Fragment layouts per CDNA5 ISA 7.12.2 (wave32), f16:
//   A 16x32: lane m=lane&15; khalf=(lane<16)?0:8;
//            vgpr j<4 -> K=khalf+2j(,+1); j>=4 -> K=16+khalf+2(j-4)(,+1)
//   B 32x16: lane n=lane&15; khalf=(lane<16)?0:8;
//            vgpr j<4 -> K=khalf+2j(,+1); j>=4 -> K=16+khalf+2(j-4)(,+1)
//   C/D 16x16 f32: vgpr r -> M = r + ((lane<16)?0:8), N = lane&15
// ---------------------------------------------------------------------------
#define GEMM_LDS_H 8192

__global__ void k_wmma_gemm(const _Float16* __restrict__ A, const float* __restrict__ bias,
                            float* __restrict__ C, TDesc td, int hasbias,
                            int M, int Kp, int Np, int Nact, int act) {
  __shared__ _Float16 Bs[GEMM_LDS_H];
  __shared__ float BiasS[64];
  int tid = threadIdx.x;
  int lane = tid & 31;
  int wid = tid >> 5;

  // Stage bias into LDS via the async-to-LDS path (ASYNCcnt).
  if (hasbias && tid < Np) {
    unsigned lo = (unsigned)(size_t)(&BiasS[tid]);
    unsigned long long ga = (unsigned long long)(size_t)(bias + tid);
    asm volatile("global_load_async_to_lds_b32 %0, %1, off" :: "v"(lo), "v"(ga) : "memory");
  }

  // Wave 0 issues one TDM load of the whole B panel into LDS (TENSORcnt).
  if (wid == 0) {
    unsigned ldsoff = (unsigned)(size_t)(&Bs[0]);
    u32x4 G0 = { td.d[0], ldsoff, td.d[2], td.d[3] };
    u32x8 G1 = { td.d[4], td.d[5], td.d[6], td.d[7], td.d[8], td.d[9], td.d[10], td.d[11] };
    asm volatile("tensor_load_to_lds %0, %1" :: "s"(G0), "s"(G1) : "memory");
    __builtin_amdgcn_s_wait_tensorcnt(0);
  }
  asm volatile("s_wait_asynccnt 0" ::: "memory");
  __syncthreads();

  int mtile = blockIdx.x * (blockDim.x >> 5) + wid;
  if (mtile >= (M >> 4)) return;   // wave-uniform
  int tilesN = Np >> 4;
  int abase = (lane < 16) ? 0 : 8;
  int bbase = (lane < 16) ? 0 : 8;
  const _Float16* arow = A + (size_t)(mtile * 16 + (lane & 15)) * Kp;
  const _Float16* bls = &Bs[(size_t)(lane & 15) * Kp];
  v8f acc0 = {}, acc1 = {}, acc2 = {}, acc3 = {};

  for (int k0 = 0; k0 < Kp; k0 += 32) {
    __builtin_prefetch(arow + k0 + 32, 0, 0);   // next A K-strip -> global_prefetch_b8
    v16h af;
#pragma unroll
    for (int j = 0; j < 8; ++j) {
      int ka = k0 + ((j < 4) ? (abase + 2 * j) : (16 + abase + 2 * (j - 4)));
      af[2 * j]     = arow[ka];
      af[2 * j + 1] = arow[ka + 1];
    }
#pragma unroll
    for (int tn = 0; tn < 4; ++tn) {
      if (tn >= tilesN) break;
      const _Float16* br = bls + (size_t)(tn * 16) * Kp;
      v16h bf;
#pragma unroll
      for (int j = 0; j < 8; ++j) {
        int kb = k0 + ((j < 4) ? (bbase + 2 * j) : (16 + bbase + 2 * (j - 4)));
        bf[2 * j]     = br[kb];
        bf[2 * j + 1] = br[kb + 1];
      }
      if      (tn == 0) acc0 = __builtin_amdgcn_wmma_f32_16x16x32_f16(false, af, false, bf, (short)0, acc0, false, false);
      else if (tn == 1) acc1 = __builtin_amdgcn_wmma_f32_16x16x32_f16(false, af, false, bf, (short)0, acc1, false, false);
      else if (tn == 2) acc2 = __builtin_amdgcn_wmma_f32_16x16x32_f16(false, af, false, bf, (short)0, acc2, false, false);
      else              acc3 = __builtin_amdgcn_wmma_f32_16x16x32_f16(false, af, false, bf, (short)0, acc3, false, false);
    }
  }

  int mb = mtile * 16 + ((lane < 16) ? 0 : 8);
#pragma unroll
  for (int tn = 0; tn < 4; ++tn) {
    if (tn >= tilesN) break;
    int n = tn * 16 + (lane & 15);
    if (n >= Nact) continue;
    float bv = hasbias ? BiasS[n] : 0.0f;
    v8f acc = (tn == 0) ? acc0 : (tn == 1) ? acc1 : (tn == 2) ? acc2 : acc3;
#pragma unroll
    for (int r = 0; r < 8; ++r) {
      float v = acc[r] + bv;
      if (act == 1) v = gelu_f(v);
      else if (act == 2) v = silu_f(v);
      else if (act == 3) v = sigm_f(v);
      C[(size_t)(mb + r) * Np + n] = v;
    }
  }
}

// im2col for 3x3 pad-1 conv: NCHW f32 -> f16 A[M=B*H*W, Kp], 8 halfs per thread.
__global__ void k_im2col3x3(const float* __restrict__ x, _Float16* __restrict__ A,
                            int Bn, int Cin, int H, int W, int Kp) {
  int Kg = Kp >> 3;
  size_t idx = (size_t)blockIdx.x * blockDim.x + threadIdx.x;
  size_t tot = (size_t)Bn * H * W * Kg;
  if (idx >= tot) return;
  int kg = (int)(idx % Kg);
  size_t m = idx / Kg;
  int xw = (int)(m % W);
  size_t r = m / W;
  int yh = (int)(r % H);
  int b = (int)(r / H);
  int K = Cin * 9;
  _Float16 vv[8];
#pragma unroll
  for (int jj = 0; jj < 8; ++jj) {
    int k = (kg << 3) + jj;
    _Float16 v = (_Float16)0.0f;
    if (k < K) {
      int ci = k / 9, t = k - ci * 9, ky = t / 3, kx = t - ky * 3;
      int yy = yh + ky - 1, xx = xw + kx - 1;
      if (yy >= 0 && yy < H && xx >= 0 && xx < W)
        v = (_Float16)x[(((size_t)b * Cin + ci) * H + yy) * W + xx];
    }
    vv[jj] = v;
  }
  *(int4*)(A + m * Kp + ((size_t)kg << 3)) = *(int4*)vv;
}

// Pack weight W[Nact,Kact] (row-major f32) into f16 Bp[Np,Kp], zero-padded.
__global__ void k_pack_mat(const float* __restrict__ Wm, _Float16* __restrict__ Bp,
                           int Nact, int Kact, int Np, int Kp) {
  int idx = blockIdx.x * blockDim.x + threadIdx.x;
  if (idx >= Np * Kp) return;
  int k = idx % Kp, n = idx / Kp;
  _Float16 v = (_Float16)0.0f;
  if (n < Nact && k < Kact) v = (_Float16)Wm[(size_t)n * Kact + k];
  Bp[idx] = v;
}

// GroupNorm stats over NHWC buffer with padded channel stride Np. One block per (b,g).
__global__ void k_gn_stats(const float* __restrict__ O, float* __restrict__ stats,
                           int HW, int C, int Np) {
  int b = blockIdx.x >> 2, g = blockIdx.x & 3;
  int Cg = C >> 2;
  __shared__ float ssum[256], ssq[256];
  float s = 0.0f, q = 0.0f;
  int total = HW * Cg;
  for (int i = threadIdx.x; i < total; i += blockDim.x) {
    int c = g * Cg + (i % Cg);
    int p = i / Cg;
    float v = O[((size_t)b * HW + p) * Np + c];
    s += v; q += v * v;
  }
  ssum[threadIdx.x] = s; ssq[threadIdx.x] = q;
  __syncthreads();
  for (int st = blockDim.x / 2; st > 0; st >>= 1) {
    if ((int)threadIdx.x < st) { ssum[threadIdx.x] += ssum[threadIdx.x + st]; ssq[threadIdx.x] += ssq[threadIdx.x + st]; }
    __syncthreads();
  }
  if (threadIdx.x == 0) {
    float m = ssum[0] / total;
    float var = ssq[0] / total - m * m;
    stats[blockIdx.x * 2] = m;
    stats[blockIdx.x * 2 + 1] = rsqrtf(var + 1e-5f);
  }
}

// gelu(maxpool2(groupnorm(NHWC))) -> NCHW [B,C,H/2,W/2]
__global__ void k_gn_pool_gelu(const float* __restrict__ O, const float* __restrict__ stats,
                               const float* __restrict__ gw, const float* __restrict__ gb,
                               float* __restrict__ out, int Bn, int C, int H, int W, int Np) {
  int Ho = H >> 1, Wo = W >> 1;
  size_t idx = (size_t)blockIdx.x * blockDim.x + threadIdx.x;
  size_t tot = (size_t)Bn * C * Ho * Wo;
  if (idx >= tot) return;
  int xo = (int)(idx % Wo);
  size_t r = idx / Wo;
  int yo = (int)(r % Ho); r /= Ho;
  int c = (int)(r % C);
  int b = (int)(r / C);
  int g = c / (C >> 2);
  float m = stats[(b * 4 + g) * 2], rs = stats[(b * 4 + g) * 2 + 1];
  float gwc = gw[c], gbc = gb[c];
  float mx = -3.4e38f;
  for (int dy = 0; dy < 2; ++dy)
    for (int dx = 0; dx < 2; ++dx) {
      float v = O[((size_t)b * H * W + (size_t)(2 * yo + dy) * W + (2 * xo + dx)) * Np + c];
      v = (v - m) * rs * gwc + gbc;
      mx = fmaxf(mx, v);
    }
  out[idx] = gelu_f(mx);
}

// groupnorm(NHWC) -> NCHW (no activation)
__global__ void k_gn_nchw(const float* __restrict__ O, const float* __restrict__ stats,
                          const float* __restrict__ gw, const float* __restrict__ gb,
                          float* __restrict__ out, int Bn, int C, int H, int W, int Np) {
  size_t idx = (size_t)blockIdx.x * blockDim.x + threadIdx.x;
  size_t tot = (size_t)Bn * C * H * W;
  if (idx >= tot) return;
  int x = (int)(idx % W);
  size_t r = idx / W;
  int y = (int)(r % H); r /= H;
  int c = (int)(r % C);
  int b = (int)(r / C);
  int g = c / (C >> 2);
  float m = stats[(b * 4 + g) * 2], rs = stats[(b * 4 + g) * 2 + 1];
  float v = O[((size_t)b * H * W + (size_t)y * W + x) * Np + c];
  out[idx] = (v - m) * rs * gw[c] + gb[c];
}

// gelu(groupnorm(NHWC)) + skip(NCHW) -> NCHW
__global__ void k_gn_gelu_add(const float* __restrict__ O, const float* __restrict__ stats,
                              const float* __restrict__ gw, const float* __restrict__ gb,
                              const float* __restrict__ skip, float* __restrict__ out,
                              int Bn, int C, int H, int W, int Np) {
  size_t idx = (size_t)blockIdx.x * blockDim.x + threadIdx.x;
  size_t tot = (size_t)Bn * C * H * W;
  if (idx >= tot) return;
  int x = (int)(idx % W);
  size_t r = idx / W;
  int y = (int)(r % H); r /= H;
  int c = (int)(r % C);
  int b = (int)(r / C);
  int g = c / (C >> 2);
  float m = stats[(b * 4 + g) * 2], rs = stats[(b * 4 + g) * 2 + 1];
  float v = O[((size_t)b * H * W + (size_t)y * W + x) * Np + c];
  v = (v - m) * rs * gw[c] + gb[c];
  out[idx] = gelu_f(v) + skip[idx];
}

// gelu(bilinear_upsample2_align_corners(NCHW)) + skip -> NCHW [B,C,2H,2W]
__global__ void k_up_gelu_add(const float* __restrict__ G, const float* __restrict__ skip,
                              float* __restrict__ out, int Bn, int C, int H, int W) {
  int Ho = 2 * H, Wo = 2 * W;
  size_t idx = (size_t)blockIdx.x * blockDim.x + threadIdx.x;
  size_t tot = (size_t)Bn * C * Ho * Wo;
  if (idx >= tot) return;
  int x = (int)(idx % Wo);
  size_t r = idx / Wo;
  int y = (int)(r % Ho); r /= Ho;
  int c = (int)(r % C);
  int b = (int)(r / C);
  float cy = y * (float)(H - 1) / (float)(Ho - 1);
  float cx = x * (float)(W - 1) / (float)(Wo - 1);
  int i0 = (int)floorf(cy); if (i0 > H - 2) i0 = H - 2;
  int j0 = (int)floorf(cx); if (j0 > W - 2) j0 = W - 2;
  float fy = cy - i0, fx = cx - j0;
  const float* p = G + (((size_t)b * C + c) * H) * W;
  float v00 = p[(size_t)i0 * W + j0],       v01 = p[(size_t)i0 * W + j0 + 1];
  float v10 = p[(size_t)(i0 + 1) * W + j0], v11 = p[(size_t)(i0 + 1) * W + j0 + 1];
  float v = (1 - fy) * ((1 - fx) * v00 + fx * v01) + fy * ((1 - fx) * v10 + fx * v11);
  out[idx] = gelu_f(v) + skip[idx];
}

// PVM front: layernorm over C, save xn[B,N,C] (f32), and scatter channel
// groups into f16 A1[(g*B+b)*N+n, 32] for the in-projection GEMM.
__global__ void k_ln_split_pack(const float* __restrict__ X, const float* __restrict__ lw,
                                const float* __restrict__ lb, float* __restrict__ xn,
                                _Float16* __restrict__ A, int Bn, int Nn, int C,
                                int chw, int ldx) {
  size_t idx = (size_t)blockIdx.x * blockDim.x + threadIdx.x;
  if (idx >= (size_t)Bn * Nn) return;
  int n = (int)(idx % Nn);
  int b = (int)(idx / Nn);
  float buf[64];
  float s = 0.0f;
  for (int c = 0; c < C; ++c) {
    float v = chw ? X[((size_t)b * C + c) * Nn + n] : X[((size_t)b * Nn + n) * ldx + c];
    buf[c] = v; s += v;
  }
  float m = s / C, q = 0.0f;
  for (int c = 0; c < C; ++c) { float d = buf[c] - m; q += d * d; }
  float rs = rsqrtf(q / C + 1e-5f);
  int c4 = C >> 2;
  for (int c = 0; c < C; ++c) {
    float v = (buf[c] - m) * rs * lw[c] + lb[c];
    xn[((size_t)b * Nn + n) * C + c] = v;
    int g = c / c4, cc = c - g * c4;
    A[((size_t)(g * Bn + b) * Nn + n) * 32 + cc] = (_Float16)v;
  }
  for (int g = 0; g < 4; ++g)
    for (int cc = c4; cc < 32; ++cc)
      A[((size_t)(g * Bn + b) * Nn + n) * 32 + cc] = (_Float16)0.0f;
}

// Causal depthwise conv1d (k=4) + SiLU over xz[:, 0:di]; writes u (f32, ld=di)
// and f16 A2[M4,32] for the x-projection GEMM. Threads: (gb, dcol<32).
__global__ void k_dwconv_silu(const float* __restrict__ xz, const float* __restrict__ cw,
                              const float* __restrict__ cb, float* __restrict__ U,
                              _Float16* __restrict__ A2, int GB, int L, int di, int ldxz) {
  int idx = blockIdx.x * blockDim.x + threadIdx.x;
  int d = idx & 31, gb = idx >> 5;
  if (gb >= GB) return;
  if (d >= di) {
    for (int l = 0; l < L; ++l) A2[((size_t)gb * L + l) * 32 + d] = (_Float16)0.0f;
    return;
  }
  float w0 = cw[d * 4 + 0], w1 = cw[d * 4 + 1], w2 = cw[d * 4 + 2], w3 = cw[d * 4 + 3];
  float bb = cb[d];
  for (int l = 0; l < L; ++l) {
    float acc = bb;
    size_t base = (size_t)gb * L;
    if (l >= 3) acc += w0 * xz[(base + l - 3) * ldxz + d];
    if (l >= 2) acc += w1 * xz[(base + l - 2) * ldxz + d];
    if (l >= 1) acc += w2 * xz[(base + l - 1) * ldxz + d];
    acc += w3 * xz[(base + l) * ldxz + d];
    float u = silu_f(acc);
    U[(base + l) * di + d] = u;
    A2[(base + l) * 32 + d] = (_Float16)u;
  }
}

// Selective scan: per (gb,d) thread, 16-state recurrence over L.
// x_dbl layout (ld=48): [0]=dt, [1..16]=B, [17..32]=C. Writes (y*silu(z)) into A3.
__global__ void k_ssm_scan(const float* __restrict__ xdbl, const float* __restrict__ U,
                           const float* __restrict__ xz, const float* __restrict__ dtw,
                           const float* __restrict__ dtb, const float* __restrict__ alog,
                           const float* __restrict__ Dp, _Float16* __restrict__ A3,
                           int GB, int L, int di, int ldxz) {
  int idx = blockIdx.x * blockDim.x + threadIdx.x;
  int d = idx & 31, gb = idx >> 5;
  if (gb >= GB) return;
  if (d >= di) {
    for (int l = 0; l < L; ++l) A3[((size_t)gb * L + l) * 32 + d] = (_Float16)0.0f;
    return;
  }
  float Ae[16], h[16];
#pragma unroll
  for (int nn = 0; nn < 16; ++nn) { Ae[nn] = -__expf(alog[d * 16 + nn]); h[nn] = 0.0f; }
  float wdt = dtw[d], bdt = dtb[d], Dd = Dp[d];
  for (int l = 0; l < L; ++l) {
    size_t m = (size_t)gb * L + l;
    const float* xd = xdbl + m * 48;
    float delta = softplus_f(xd[0] * wdt + bdt);
    float uv = U[m * di + d];
    float du = delta * uv;
    float y = 0.0f;
#pragma unroll
    for (int nn = 0; nn < 16; ++nn) {
      h[nn] = __expf(delta * Ae[nn]) * h[nn] + du * xd[1 + nn];
      y += h[nn] * xd[17 + nn];
    }
    y += uv * Dd;
    float zv = xz[m * ldxz + di + d];
    A3[m * 32 + d] = (_Float16)(y * silu_f(zv));
  }
}

// PVM back: gather ym groups + skip*xn, layernorm, pack f16 A4[MB,Kp2].
__global__ void k_unsplit_ln_pack(const float* __restrict__ ym, const float* __restrict__ xn,
                                  const float* __restrict__ skipp, const float* __restrict__ lw,
                                  const float* __restrict__ lb, _Float16* __restrict__ A4,
                                  int Bn, int Nn, int C, int Kp2) {
  size_t idx = (size_t)blockIdx.x * blockDim.x + threadIdx.x;
  if (idx >= (size_t)Bn * Nn) return;
  int n = (int)(idx % Nn);
  int b = (int)(idx / Nn);
  int c4 = C >> 2;
  float sk = skipp[0];
  float buf[64];
  float s = 0.0f;
  for (int c = 0; c < C; ++c) {
    int g = c / c4, cc = c - g * c4;
    float v = ym[((size_t)(g * Bn + b) * Nn + n) * 16 + cc] +
              sk * xn[((size_t)b * Nn + n) * C + c];
    buf[c] = v; s += v;
  }
  float m = s / C, q = 0.0f;
  for (int c = 0; c < C; ++c) { float d = buf[c] - m; q += d * d; }
  float rs = rsqrtf(q / C + 1e-5f);
  for (int c = 0; c < C; ++c)
    A4[idx * Kp2 + c] = (_Float16)((buf[c] - m) * rs * lw[c] + lb[c]);
  for (int c = C; c < Kp2; ++c) A4[idx * Kp2 + c] = (_Float16)0.0f;
}

// ---- SC bridge kernels ----
__global__ void k_meanmax(const float* __restrict__ t, float* __restrict__ amap,
                          int Bn, int C, int H, int W) {
  size_t idx = (size_t)blockIdx.x * blockDim.x + threadIdx.x;
  if (idx >= (size_t)Bn * H * W) return;
  int x = (int)(idx % W);
  size_t r = idx / W;
  int y = (int)(r % H);
  int b = (int)(r / H);
  float s = 0.0f, mx = -3.4e38f;
  for (int c = 0; c < C; ++c) {
    float v = t[(((size_t)b * C + c) * H + y) * W + x];
    s += v; mx = fmaxf(mx, v);
  }
  amap[(((size_t)b * 2 + 0) * H + y) * W + x] = s / C;
  amap[(((size_t)b * 2 + 1) * H + y) * W + x] = mx;
}

// spatial attention: sigmoid(conv7x7 dil3 pad9 over [mean,max]) * t -> s
__global__ void k_satt(const float* __restrict__ amap, const float* __restrict__ t,
                       const float* __restrict__ sw, const float* __restrict__ sb,
                       float* __restrict__ s, int Bn, int C, int H, int W) {
  size_t idx = (size_t)blockIdx.x * blockDim.x + threadIdx.x;
  if (idx >= (size_t)Bn * H * W) return;
  int x = (int)(idx % W);
  size_t r = idx / W;
  int y = (int)(r % H);
  int b = (int)(r / H);
  float acc = sb[0];
  for (int ch = 0; ch < 2; ++ch)
    for (int ky = 0; ky < 7; ++ky) {
      int yy = y + 3 * (ky - 3);
      if (yy < 0 || yy >= H) continue;
      for (int kx = 0; kx < 7; ++kx) {
        int xx = x + 3 * (kx - 3);
        if (xx < 0 || xx >= W) continue;
        acc += sw[ch * 49 + ky * 7 + kx] * amap[(((size_t)b * 2 + ch) * H + yy) * W + xx];
      }
    }
  float a = sigm_f(acc);
  for (int c = 0; c < C; ++c) {
    size_t o = (((size_t)b * C + c) * H + y) * W + x;
    s[o] = a * t[o];
  }
}

__global__ void k_add(const float* __restrict__ a, const float* __restrict__ b,
                      float* __restrict__ o, size_t n) {
  size_t idx = (size_t)blockIdx.x * blockDim.x + threadIdx.x;
  if (idx < n) o[idx] = a[idx] + b[idx];
}

__global__ void k_chanmean(const float* __restrict__ u, float* __restrict__ attc,
                           int Bn, int C, int H, int W, int off) {
  int idx = blockIdx.x * blockDim.x + threadIdx.x;
  if (idx >= Bn * C) return;
  int c = idx % C, b = idx / C;
  float s = 0.0f;
  const float* p = u + (((size_t)b * C + c) * H) * W;
  for (int i = 0; i < H * W; ++i) s += p[i];
  attc[b * 128 + off + c] = s / (H * W);
}

__global__ void k_catt(const float* __restrict__ attc, const float* __restrict__ cw,
                       float* __restrict__ attf, int Bn) {
  int idx = blockIdx.x * blockDim.x + threadIdx.x;
  if (idx >= Bn * 128) return;
  int j = idx % 128, b = idx / 128;
  float s = 0.0f;
  for (int t = 0; t < 3; ++t) {
    int jj = j + t - 1;
    if (jj >= 0 && jj < 128) s += cw[t] * attc[b * 128 + jj];
  }
  attf[idx] = s;
}

__global__ void k_gate(const float* __restrict__ attf, const float* __restrict__ aw,
                       const float* __restrict__ ab, float* __restrict__ gates,
                       int Bn, int C) {
  int idx = blockIdx.x * blockDim.x + threadIdx.x;
  if (idx >= Bn * C) return;
  int c = idx % C, b = idx / C;
  float s = ab[c];
  for (int j = 0; j < 128; ++j) s += attf[b * 128 + j] * aw[c * 128 + j];
  gates[b * 64 + c] = sigm_f(s);
}

__global__ void k_apply(const float* __restrict__ gates, const float* __restrict__ u,
                        const float* __restrict__ s, float* __restrict__ o,
                        int Bn, int C, int H, int W) {
  size_t idx = (size_t)blockIdx.x * blockDim.x + threadIdx.x;
  size_t tot = (size_t)Bn * C * H * W;
  if (idx >= tot) return;
  size_t r = idx / ((size_t)H * W);
  int c = (int)(r % C), b = (int)(r / C);
  o[idx] = gates[b * 64 + c] * u[idx] + s[idx];
}

__global__ void k_conv1x1(const float* __restrict__ in, const float* __restrict__ fw,
                          const float* __restrict__ fb, float* __restrict__ out,
                          int Bn, int C, int H, int W) {
  size_t idx = (size_t)blockIdx.x * blockDim.x + threadIdx.x;
  if (idx >= (size_t)Bn * H * W) return;
  int x = (int)(idx % W);
  size_t r = idx / W;
  int y = (int)(r % H);
  int b = (int)(r / H);
  float acc = fb[0];
  for (int c = 0; c < C; ++c) acc += fw[c] * in[(((size_t)b * C + c) * H + y) * W + x];
  out[idx] = acc;
}

__global__ void k_up_sigmoid(const float* __restrict__ f1, float* __restrict__ out,
                             int Bn, int H, int W) {
  int Ho = 2 * H, Wo = 2 * W;
  size_t idx = (size_t)blockIdx.x * blockDim.x + threadIdx.x;
  size_t tot = (size_t)Bn * Ho * Wo;
  if (idx >= tot) return;
  int x = (int)(idx % Wo);
  size_t r = idx / Wo;
  int y = (int)(r % Ho);
  int b = (int)(r / Ho);
  float cy = y * (float)(H - 1) / (float)(Ho - 1);
  float cx = x * (float)(W - 1) / (float)(Wo - 1);
  int i0 = (int)floorf(cy); if (i0 > H - 2) i0 = H - 2;
  int j0 = (int)floorf(cx); if (j0 > W - 2) j0 = W - 2;
  float fy = cy - i0, fx = cx - j0;
  const float* p = f1 + (size_t)b * H * W;
  float v = (1 - fy) * ((1 - fx) * p[(size_t)i0 * W + j0] + fx * p[(size_t)i0 * W + j0 + 1]) +
            fy * ((1 - fx) * p[(size_t)(i0 + 1) * W + j0] + fx * p[(size_t)(i0 + 1) * W + j0 + 1]);
  out[idx] = sigm_f(v);
}

// ---------------------------------------------------------------------------
// Host orchestration
// ---------------------------------------------------------------------------
// Input index map (setup_inputs insertion order):
//  0:x 1:e1w 2:e1b 3:e2w 4:e2b 5:e3w 6:e3b 7:d4w 8:d4b 9:d5w 10:d5b 11:fw 12:fb
//  13..22: ebn{1..5}{w,b}  23..32: dbn{1..5}{w,b}
//  pvm blocks (14 entries each): pvm4=33 pvm5=47 pvm6=61 pvmd1=75 pvmd2=89 pvmd3=103
//  117:satt_w 118:satt_b 119:catt_w 120..124:att_w[0..4] 125..129:att_b[0..4]
enum { P_LNW = 0, P_LNB, P_PROJW, P_PROJB, P_SKIP, P_INW, P_CONVW, P_CONVB,
       P_XPROJW, P_DTW, P_DTB, P_ALOG, P_D, P_OUTW };

extern "C" void kernel_launch(void* const* d_in, const int* in_sizes, int n_in,
                              void* d_out, int out_size, void* d_ws, size_t ws_size,
                              hipStream_t stream) {
  (void)in_sizes; (void)n_in; (void)out_size; (void)ws_size;
  const int B = 32;
  auto P = [&](int i) -> const float* { return (const float*)d_in[i]; };

  // workspace bump allocator
  char* wp = (char*)d_ws;
  size_t off = 0;
  auto alloc = [&](size_t bytes) -> void* {
    void* p = wp + off;
    off = (off + bytes + 255) & ~(size_t)255;
    return p;
  };
  auto allocF = [&](size_t n) -> float* { return (float*)alloc(n * 4); };
  auto allocH = [&](size_t n) -> _Float16* { return (_Float16*)alloc(n * 2); };

  _Float16* A16 = allocH(67108864);      // im2col arena (e1: 2M x 32)
  _Float16* B16 = allocH(32768);         // packed weights
  float*    O32 = allocF(33554432);      // GEMM output arena (e1: 2M x 16)
  _Float16* AA = allocH(4194304);        // pvm A1
  _Float16* AB = allocH(4194304);        // pvm A2
  _Float16* AC = allocH(4194304);        // pvm A3
  _Float16* AD = allocH(2097152);        // pvm A4
  float* G1 = allocF(1048576);           // xn
  float* G2 = allocF(4194304);           // xz
  float* G3 = allocF(6291456);           // x_dbl
  float* G4 = allocF(4194304);           // u
  float* G5 = allocF(2097152);           // ym
  float* stats = allocF(256);
  float* t1 = allocF(4194304); float* t2 = allocF(2097152); float* t3 = allocF(786432);
  float* t4 = allocF(262144);  float* t5 = allocF(98304);
  float* s1 = allocF(4194304); float* s2 = allocF(2097152); float* s3 = allocF(786432);
  float* s4 = allocF(262144);  float* s5 = allocF(98304);
  float* u1 = allocF(4194304); float* u2 = allocF(2097152); float* u3 = allocF(786432);
  float* u4 = allocF(262144);  float* u5 = allocF(98304);
  float* o1 = allocF(4194304); float* o2 = allocF(2097152); float* o3 = allocF(786432);
  float* o4 = allocF(262144);  float* o5 = allocF(98304);
  float* amap = allocF(1048576);
  float* attc = allocF(4096); float* attf = allocF(4096); float* gates = allocF(2048);
  float* gbuf = allocF(1048576);
  float* out5 = allocF(98304); float* out4 = allocF(262144); float* out3 = allocF(786432);
  float* out2 = allocF(2097152); float* out1 = allocF(4194304);
  float* pvm6o = allocF(131072);
  float* f1 = allocF(524288);

  // Build the TDM descriptor for a B panel of `total` f16 elements at `Bm`:
  // 2D tensor (dim0=total, dim1=1), one tile covering the whole panel.
  auto make_td = [&](const _Float16* Bm, unsigned total) -> TDesc {
    unsigned long long ga = (unsigned long long)(size_t)Bm;
    TDesc t;
    t.d[0] = 1u;                                              // count=1
    t.d[1] = 0u;                                              // lds_addr (kernel patches)
    t.d[2] = (unsigned)(ga & 0xFFFFFFFFu);                    // global_addr[31:0]
    t.d[3] = (unsigned)((ga >> 32) & 0x01FFFFFFu) | 0x80000000u; // addr[56:32] | type=2
    t.d[4] = 1u << 16;                                        // data_size=2B
    t.d[5] = (total & 0xFFFFu) << 16;                         // tensor_dim0[15:0]
    t.d[6] = (total >> 16) | (1u << 16);                      // tensor_dim0[31:16] | tensor_dim1=1
    t.d[7] = total << 16;                                     // tile_dim0=total (<=5120)
    t.d[8] = 1u;                                              // tile_dim1=1
    t.d[9] = total;                                           // tensor_dim0_stride
    t.d[10] = 0u;
    t.d[11] = 0u;
    return t;
  };

  auto gemm = [&](const _Float16* A, const _Float16* Bm, float* C, const float* bias,
                  int M, int Kp, int Np, int Nact, int act) {
    int tilesM = M / 16;
    TDesc td = make_td(Bm, (unsigned)(Np * Kp));
    k_wmma_gemm<<<CDIV(tilesM, 8), 256, 0, stream>>>(A, bias, C, td, bias != nullptr,
                                                     M, Kp, Np, Nact, act);
  };
  auto gnstats = [&](const float* O, int HW, int C, int Np) {
    k_gn_stats<<<B * 4, 256, 0, stream>>>(O, stats, HW, C, Np);
  };
  auto conv_gemm = [&](const float* in, int Cin, int Cout, int H, int W,
                       const float* w, const float* bias) -> int {
    int Kc = Cin * 9, Kp = CDIV(Kc, 32) * 32, Np = CDIV(Cout, 16) * 16;
    size_t M = (size_t)B * H * W;
    k_im2col3x3<<<(int)CDIV(M * (Kp / 8), 256), 256, 0, stream>>>(in, A16, B, Cin, H, W, Kp);
    k_pack_mat<<<CDIV(Np * Kp, 256), 256, 0, stream>>>(w, B16, Cout, Kc, Np, Kp);
    gemm(A16, B16, O32, bias, (int)M, Kp, Np, Cout, 0);
    return Np;
  };
  auto enc_conv = [&](const float* in, int Cin, int Cout, int H, int W,
                      const float* w, const float* bias, const float* gw,
                      const float* gb, float* out) {
    int Np = conv_gemm(in, Cin, Cout, H, W, w, bias);
    gnstats(O32, H * W, Cout, Np);
    size_t tot = (size_t)B * Cout * (H / 2) * (W / 2);
    k_gn_pool_gelu<<<(int)CDIV(tot, 256), 256, 0, stream>>>(O32, stats, gw, gb, out, B, Cout, H, W, Np);
  };

  // in: NCHW (chw=1) or NHWC with leading dim ldx (chw=0); out: NHWC f32 [B*N, NpProj]
  auto run_pvm = [&](const float* in, int chw, int ldx, int H, int W, int din, int dout,
                     int pb, int act, float* outNHWC) -> int {
    int dm = din / 4, di = 2 * dm, L = H * W;
    int MB = B * L, M4 = 4 * MB;
    int NpIn = CDIV(2 * di, 16) * 16;
    int Kp2 = (din <= 32) ? 32 : 64;
    int NpProj = CDIV(dout, 16) * 16;
    k_ln_split_pack<<<CDIV(MB, 256), 256, 0, stream>>>(in, P(pb + P_LNW), P(pb + P_LNB),
                                                       G1, AA, B, L, din, chw, ldx);
    k_pack_mat<<<CDIV(NpIn * 32, 256), 256, 0, stream>>>(P(pb + P_INW), B16, 2 * di, dm, NpIn, 32);
    gemm(AA, B16, G2, nullptr, M4, 32, NpIn, 2 * di, 0);
    k_dwconv_silu<<<CDIV(4 * B * 32, 256), 256, 0, stream>>>(G2, P(pb + P_CONVW), P(pb + P_CONVB),
                                                             G4, AB, 4 * B, L, di, NpIn);
    k_pack_mat<<<CDIV(48 * 32, 256), 256, 0, stream>>>(P(pb + P_XPROJW), B16, 33, di, 48, 32);
    gemm(AB, B16, G3, nullptr, M4, 32, 48, 33, 0);
    k_ssm_scan<<<CDIV(4 * B * 32, 256), 256, 0, stream>>>(G3, G4, G2, P(pb + P_DTW), P(pb + P_DTB),
                                                          P(pb + P_ALOG), P(pb + P_D), AC, 4 * B, L, di, NpIn);
    k_pack_mat<<<CDIV(16 * 32, 256), 256, 0, stream>>>(P(pb + P_OUTW), B16, dm, di, 16, 32);
    gemm(AC, B16, G5, nullptr, M4, 32, 16, dm, 0);
    k_unsplit_ln_pack<<<CDIV(MB, 256), 256, 0, stream>>>(G5, G1, P(pb + P_SKIP), P(pb + P_LNW),
                                                         P(pb + P_LNB), AD, B, L, din, Kp2);
    k_pack_mat<<<CDIV(NpProj * Kp2, 256), 256, 0, stream>>>(P(pb + P_PROJW), B16, dout, din, NpProj, Kp2);
    gemm(AD, B16, outNHWC, P(pb + P_PROJB), MB, Kp2, NpProj, dout, act);
    return NpProj;
  };

  // ---------------- Encoder ----------------
  enc_conv(P(0), 3, 8, 256, 256, P(1), P(2), P(13), P(14), t1);    // t1: [B,8,128,128]
  enc_conv(t1, 8, 16, 128, 128, P(3), P(4), P(15), P(16), t2);     // t2: [B,16,64,64]
  enc_conv(t2, 16, 24, 64, 64, P(5), P(6), P(17), P(18), t3);      // t3: [B,24,32,32]

  { // pvm4 -> gn(ebn4) -> pool -> gelu -> t4
    int Np = run_pvm(t3, 1, 0, 32, 32, 24, 32, 33, 0, O32);
    gnstats(O32, 1024, 32, Np);
    size_t tot = (size_t)B * 32 * 16 * 16;
    k_gn_pool_gelu<<<(int)CDIV(tot, 256), 256, 0, stream>>>(O32, stats, P(19), P(20), t4, B, 32, 32, 32, Np);
  }
  { // pvm5 -> gn(ebn5) -> pool -> gelu -> t5
    int Np = run_pvm(t4, 1, 0, 16, 16, 32, 48, 47, 0, O32);
    gnstats(O32, 256, 48, Np);
    size_t tot = (size_t)B * 48 * 8 * 8;
    k_gn_pool_gelu<<<(int)CDIV(tot, 256), 256, 0, stream>>>(O32, stats, P(21), P(22), t5, B, 48, 16, 16, Np);
  }

  // ---------------- SC bridge ----------------
  {
    float* T[5] = {t1, t2, t3, t4, t5};
    float* S[5] = {s1, s2, s3, s4, s5};
    float* U[5] = {u1, u2, u3, u4, u5};
    float* O[5] = {o1, o2, o3, o4, o5};
    int Cc[5] = {8, 16, 24, 32, 48};
    int Hh[5] = {128, 64, 32, 16, 8};
    int offc[5] = {0, 8, 24, 48, 80};
    for (int i = 0; i < 5; ++i) {
      int C = Cc[i], H = Hh[i], W = Hh[i];
      size_t pix = (size_t)B * H * W, tot = pix * C;
      k_meanmax<<<(int)CDIV(pix, 256), 256, 0, stream>>>(T[i], amap, B, C, H, W);
      k_satt<<<(int)CDIV(pix, 256), 256, 0, stream>>>(amap, T[i], P(117), P(118), S[i], B, C, H, W);
      k_add<<<(int)CDIV(tot, 256), 256, 0, stream>>>(S[i], T[i], U[i], tot);
      k_chanmean<<<CDIV(B * C, 64), 64, 0, stream>>>(U[i], attc, B, C, H, W, offc[i]);
    }
    k_catt<<<CDIV(B * 128, 128), 128, 0, stream>>>(attc, P(119), attf, B);
    for (int i = 0; i < 5; ++i) {
      int C = Cc[i], H = Hh[i], W = Hh[i];
      size_t tot = (size_t)B * C * H * W;
      k_gate<<<CDIV(B * C, 64), 64, 0, stream>>>(attf, P(120 + i), P(125 + i), gates, B, C);
      k_apply<<<(int)CDIV(tot, 256), 256, 0, stream>>>(gates, U[i], S[i], O[i], B, C, H, W);
    }
  }

  // ---------------- Decoder ----------------
  run_pvm(t5, 1, 0, 8, 8, 48, 64, 61, 1 /*gelu*/, pvm6o);          // NHWC [B*64, 64]
  { // pvmd1 -> gelu(gn(dbn1)) + o5 -> out5 NCHW [B,48,8,8]
    int Np = run_pvm(pvm6o, 0, 64, 8, 8, 64, 48, 75, 0, O32);
    gnstats(O32, 64, 48, Np);
    size_t tot = (size_t)B * 48 * 8 * 8;
    k_gn_gelu_add<<<(int)CDIV(tot, 256), 256, 0, stream>>>(O32, stats, P(23), P(24), o5, out5, B, 48, 8, 8, Np);
  }
  { // pvmd2 -> gn(dbn2) -> up -> gelu + o4 -> out4 [B,32,16,16]
    int Np = run_pvm(out5, 1, 0, 8, 8, 48, 32, 89, 0, O32);
    gnstats(O32, 64, 32, Np);
    size_t tg = (size_t)B * 32 * 8 * 8;
    k_gn_nchw<<<(int)CDIV(tg, 256), 256, 0, stream>>>(O32, stats, P(25), P(26), gbuf, B, 32, 8, 8, Np);
    size_t to = (size_t)B * 32 * 16 * 16;
    k_up_gelu_add<<<(int)CDIV(to, 256), 256, 0, stream>>>(gbuf, o4, out4, B, 32, 8, 8);
  }
  { // pvmd3 -> gn(dbn3) -> up -> gelu + o3 -> out3 [B,24,32,32]
    int Np = run_pvm(out4, 1, 0, 16, 16, 32, 24, 103, 0, O32);
    gnstats(O32, 256, 24, Np);
    size_t tg = (size_t)B * 24 * 16 * 16;
    k_gn_nchw<<<(int)CDIV(tg, 256), 256, 0, stream>>>(O32, stats, P(27), P(28), gbuf, B, 24, 16, 16, Np);
    size_t to = (size_t)B * 24 * 32 * 32;
    k_up_gelu_add<<<(int)CDIV(to, 256), 256, 0, stream>>>(gbuf, o3, out3, B, 24, 16, 16);
  }
  { // d4 conv -> gn(dbn4) -> up -> gelu + o2 -> out2 [B,16,64,64]
    int Np = conv_gemm(out3, 24, 16, 32, 32, P(7), P(8));
    gnstats(O32, 1024, 16, Np);
    size_t tg = (size_t)B * 16 * 32 * 32;
    k_gn_nchw<<<(int)CDIV(tg, 256), 256, 0, stream>>>(O32, stats, P(29), P(30), gbuf, B, 16, 32, 32, Np);
    size_t to = (size_t)B * 16 * 64 * 64;
    k_up_gelu_add<<<(int)CDIV(to, 256), 256, 0, stream>>>(gbuf, o2, out2, B, 16, 32, 32);
  }
  { // d5 conv -> gn(dbn5) -> up -> gelu + o1 -> out1 [B,8,128,128]
    int Np = conv_gemm(out2, 16, 8, 64, 64, P(9), P(10));
    gnstats(O32, 4096, 8, Np);
    size_t tg = (size_t)B * 8 * 64 * 64;
    k_gn_nchw<<<(int)CDIV(tg, 256), 256, 0, stream>>>(O32, stats, P(31), P(32), gbuf, B, 8, 64, 64, Np);
    size_t to = (size_t)B * 8 * 128 * 128;
    k_up_gelu_add<<<(int)CDIV(to, 256), 256, 0, stream>>>(gbuf, o1, out1, B, 8, 64, 64);
  }
  { // final 1x1 conv -> up -> sigmoid -> d_out [B,1,256,256]
    size_t pix = (size_t)B * 128 * 128;
    k_conv1x1<<<(int)CDIV(pix, 256), 256, 0, stream>>>(out1, P(11), P(12), f1, B, 8, 128, 128);
    size_t to = (size_t)B * 256 * 256;
    k_up_sigmoid<<<(int)CDIV(to, 256), 256, 0, stream>>>(f1, (float*)d_out, B, 128, 128);
  }
}
